// ParallelCNNConditionalDistribution_36249523978546
// MI455X (gfx1250) — matrix operand order repaired and checked by reading, not compiled, not run on hardware
//
#include <hip/hip_runtime.h>
#include <hip/hip_bf16.h>

typedef __attribute__((ext_vector_type(16))) _Float16 v16h;
typedef __attribute__((ext_vector_type(8)))  _Float16 v8h;
typedef __attribute__((ext_vector_type(4)))  _Float16 v4h;
typedef __attribute__((ext_vector_type(8)))  float    v8f;
typedef __attribute__((ext_vector_type(4)))  unsigned int v4u;
typedef __attribute__((ext_vector_type(8)))  int      v8i;
typedef __attribute__((ext_vector_type(4)))  int      v4i;

#define NSUB 12
#define BATCH 16
#define NB (NSUB*BATCH)   // 192 images
#define HH 128
#define WW 128

#if __has_builtin(__builtin_amdgcn_tensor_load_to_lds)
#define HAVE_TDM 1
#else
#define HAVE_TDM 0
#endif

__device__ __forceinline__ void wait_tensorcnt0() {
#if __has_builtin(__builtin_amdgcn_s_wait_tensorcnt)
  __builtin_amdgcn_s_wait_tensorcnt(0);
#else
  asm volatile("s_wait_tensorcnt 0x0" ::: "memory");
#endif
}

#if HAVE_TDM
// ---------------------------------------------------------------------------
// TDM: DMA one contiguous row of `nelem` f16 elements from global to LDS.
// D# built per CDNA5 ISA 8.3/8.4: group0 = {count=1, lds_addr, global_addr,
// type=2}; group1 = {data_size=1(2B), tensor_dim0=nelem, tensor_dim1=1,
// tile_dim0=nelem, tile_dim1=1, tensor_dim0_stride=nelem}. 2D-capable form
// (groups 2/3 zero). Tracked by TENSORcnt.
// ---------------------------------------------------------------------------
__device__ __forceinline__ void tdm_load_row(const _Float16* gsrc,
                                             unsigned ldsaddr, unsigned nelem) {
  const unsigned long long ga = (unsigned long long)(size_t)gsrc;
  const v4u g0 = { 1u,                                    // count=1 (valid D#)
                   ldsaddr,                               // lds_addr
                   (unsigned)ga,                          // global_addr[31:0]
                   (unsigned)(((ga >> 32) & 0x01ffffffu)  // global_addr[56:32]
                              | (2u << 30)) };            // type=2 ("image")
  v8i g1;
  g1[0] = 0x00010000;                                   // data_size = 2 bytes
  g1[1] = (int)((nelem & 0xffffu) << 16);               // tensor_dim0[15:0]
  g1[2] = (int)(((nelem >> 16) & 0xffffu) | (1u << 16));// dim0[31:16] | dim1=1
  g1[3] = (int)(nelem << 16);                           // tile_dim0 (<=8192)
  g1[4] = 1;                                            // tile_dim1 = 1
  g1[5] = (int)nelem;                                   // dim0_stride[31:0]
  g1[6] = 0;
  g1[7] = 0;
  const v4i zz = { 0, 0, 0, 0 };
#if defined(__clang_major__) && (__clang_major__ >= 23)
  const v8i z8 = { 0, 0, 0, 0, 0, 0, 0, 0 };
  __builtin_amdgcn_tensor_load_to_lds(g0, g1, zz, zz, z8, 0);
#else
  __builtin_amdgcn_tensor_load_to_lds(g0, g1, zz, zz, 0);
#endif
}
#endif  // HAVE_TDM

// fast tanh via exp2 (VALU trans path)
__device__ __forceinline__ float tanh_fast(float x) {
  float e = __builtin_amdgcn_exp2f(x * 2.8853900817779268f);  // exp(2x)
  return 1.0f - 2.0f * __builtin_amdgcn_rcpf(e + 1.0f);
}

// ---------------------------------------------------------------------------
// constexpr B-fragment offset helpers (half-element offsets inside the LDS
// row buffer, excluding the px*IC term). Return -1 for K-padding.
// ---------------------------------------------------------------------------
template<int IC, int KH>
__device__ constexpr int boff16(int chunk, int grp) {   // IC >= 16: 16-half run
  constexpr int Wp = WW + 2*(KH/2);
  constexpr int T  = KH*KH;
  const int k0  = chunk*32 + grp*16;
  const int tap = k0 / IC;
  if (tap >= T) return -1;
  return ((tap/KH)*Wp + (tap%KH))*IC + (k0 % IC);
}
template<int IC, int KH>
__device__ constexpr int boff4(int chunk, int grp, int q) { // IC==4: 4-half runs
  constexpr int Wp = WW + 2*(KH/2);
  constexpr int T  = KH*KH;
  const int k   = chunk*32 + grp*16 + q*4;
  const int tap = k / IC;
  if (tap >= T) return -1;
  return ((tap/KH)*Wp + (tap%KH))*IC + (k % IC);
}

// ---------------------------------------------------------------------------
// Pack f32 weights [N, OC, IC, KH, KW] into CDNA5 A-fragment layout.
// ---------------------------------------------------------------------------
template<int IC, int OC, int KH, int KC>
__global__ void pack_w(const float* __restrict__ w, _Float16* __restrict__ wp) {
  constexpr int MT = OC / 16;
  constexpr int T  = KH * KH;
  const int total = NSUB * MT * KC * 512;
  for (int i = blockIdx.x * 256 + threadIdx.x; i < total; i += gridDim.x * 256) {
    const int h     = i & 15;
    const int lane  = (i >> 4) & 31;
    const int rest  = i >> 9;
    const int chunk = rest % KC;
    const int mt    = (rest / KC) % MT;
    const int n     = rest / (KC * MT);
    const int oc    = mt * 16 + (lane & 15);
    const int k     = chunk * 32 + (lane >> 4) * 16 + h;
    const int tap   = k / IC;
    const int ic    = k % IC;
    float v = 0.0f;
    if (tap < T)
      v = w[((size_t)(n * OC + oc) * IC + ic) * T + tap];
    wp[i] = (_Float16)v;
  }
}

// ---------------------------------------------------------------------------
// Build network input xin[img][y][x][4] (f16 NHWC, img = n*16+b)
// ---------------------------------------------------------------------------
__global__ void prep_input(const float* __restrict__ samples,
                           const float* __restrict__ cond,
                           _Float16* __restrict__ xin) {
  const int i = blockIdx.x * 256 + threadIdx.x;     // over NB*H*W
  if (i >= NB * HH * WW) return;
  const int x = i & (WW - 1);
  const int y = (i >> 7) & (HH - 1);
  const int img = i >> 14;
  const int n = img >> 4, b = img & 15;
  const int p = (y & 1) ? ((x & 1) ? 1 : 3) : ((x & 1) ? 2 : 0);
  v4h o = {};
#pragma unroll
  for (int c = 0; c < 3; ++c) {
    const float s = samples[(((b * 3 + c) * HH) + y) * WW + x];
    o[c] = (_Float16)((p * 3 + c < n) ? s : 0.0f);
  }
  o[3] = (_Float16)cond[((b * HH) + y) * WW + x];
  *(v4h*)&xin[(size_t)i * 4] = o;
}

// ---------------------------------------------------------------------------
// Implicit-GEMM conv layer, WMMA f16->f32, fused bias+tanh, NHWC f16 I/O.
// One block per (image, output row). 8 waves; wave w owns px strip
// [w*16, w*16+16) and ALL M-tiles (acc[MT]); B loaded once per chunk.
// Rows staged via TDM (tensor_load_to_lds) when available; halo columns and
// out-of-range rows zero-filled with DS stores (disjoint from TDM writes).
// ---------------------------------------------------------------------------
template<int IC, int OC, int KH, int KC>
__global__ __launch_bounds__(256)
void conv_wmma(const _Float16* __restrict__ in,
               const _Float16* __restrict__ wpack,
               const float*    __restrict__ bias,
               _Float16*       __restrict__ out) {
  constexpr int PAD = KH / 2;
  constexpr int Wp  = WW + 2 * PAD;
  constexpr int MT  = OC / 16;
  constexpr int RH  = KH * Wp * IC;              // row-buffer halves
  constexpr bool STW = ((size_t)MT * KC * 512 + RH + 16) * 2 <= 64 * 1024;
  constexpr int WH   = STW ? MT * KC * 512 : 0;  // staged weight halves
  constexpr int SW_R = WH;                       // rows base
  constexpr int SW_Z = WH + RH;                  // 16-half zero pad base

  __shared__ _Float16 smem[SW_Z + 16];

  const int img  = blockIdx.x >> 7;
  const int y    = blockIdx.x & (HH - 1);
  const int tid  = threadIdx.x;
  const int lane = tid & 31;
  const int wave = tid >> 5;
  const int grp  = lane >> 4;
  const int npx  = lane & 15;
  const int sub  = img >> 4;                     // subnet id
  const _Float16* wl = wpack + (size_t)sub * (MT * KC * 512);

  // ---- stage weights (contiguous 16B copies) ----
  if constexpr (STW) {
    for (int u = tid * 8; u < WH; u += 256 * 8)
      *(v8h*)&smem[u] = *(const v8h*)&wl[u];
  }

  // ---- stage KH input rows (x halo, zero padded) ----
#if HAVE_TDM
  if constexpr (PAD > 0) {       // zero the two halo columns of every row
    for (int u = tid; u < KH * 2 * (IC / 4); u += 256) {
      const int r    = u / (2 * (IC / 4));
      const int rem  = u % (2 * (IC / 4));
      const int side = rem / (IC / 4);
      const int icq  = rem % (IC / 4);
      const int xi   = side ? (Wp - 1) : 0;
      v4h z = {};
      *(v4h*)&smem[SW_R + (r * Wp + xi) * IC + icq * 4] = z;
    }
  }
#pragma unroll
  for (int r = 0; r < KH; ++r) {                 // zero out-of-range rows
    const int row = y + r - PAD;
    if (row < 0 || row >= HH) {
      for (int u = tid * 4; u < Wp * IC; u += 256 * 4) {
        v4h z = {};
        *(v4h*)&smem[SW_R + r * Wp * IC + u] = z;
      }
    }
  }
  if (wave == 0) {                               // one wave drives the TDM
#pragma unroll
    for (int r = 0; r < KH; ++r) {
      const int row = y + r - PAD;
      if (row >= 0 && row < HH) {
        const _Float16* g = &in[(((size_t)img * HH + row) * WW) * IC];
        const unsigned ldsaddr =
            (unsigned)(size_t)&smem[SW_R + (r * Wp + PAD) * IC];
        tdm_load_row(g, ldsaddr, WW * IC);
      }
    }
    wait_tensorcnt0();
  }
#else
  {
    constexpr int UH = (IC >= 8) ? 8 : 4;        // halves per copy
    for (int u = tid; u < RH / UH; u += 256) {
      const int hidx = u * UH;
      const int ic = hidx % IC;
      const int xi = (hidx / IC) % Wp;
      const int r  = hidx / (IC * Wp);
      const int col = xi - PAD;
      const int row = y + r - PAD;
      const bool ok = (col >= 0 && col < WW && row >= 0 && row < HH);
      const _Float16* src = &in[(((size_t)img * HH + row) * WW + col) * IC + ic];
      if constexpr (UH == 8) {
        v8h v = {};
        if (ok) v = *(const v8h*)src;
        *(v8h*)&smem[SW_R + hidx] = v;
      } else {
        v4h v = {};
        if (ok) v = *(const v4h*)src;
        *(v4h*)&smem[SW_R + hidx] = v;
      }
    }
  }
#endif
  if (tid < 16) smem[SW_Z + tid] = (_Float16)0;  // zero pad block
  __syncthreads();

  v8f acc[MT];
#pragma unroll
  for (int t = 0; t < MT; ++t) { v8f z = {}; acc[t] = z; }

  const int px   = wave * 16 + npx;
  const int pxIC = SW_R + px * IC;               // runtime part of B address
  const int lane16 = lane * 16;

#pragma unroll
  for (int chunk = 0; chunk < KC; ++chunk) {
    // ---- B fragment: 16 halves per lane ----
    union { v16h v; v8h h2[2]; v4h q4[4]; } b;
    if constexpr (IC >= 16) {
      const int c0 = boff16<IC, KH>(chunk, 0);
      const int c1 = boff16<IC, KH>(chunk, 1);
      const int o0 = (c0 < 0) ? SW_Z : (pxIC + c0);
      const int o1 = (c1 < 0) ? SW_Z : (pxIC + c1);
      const int off = grp ? o1 : o0;             // one v_cndmask
      b.h2[0] = *(const v8h*)&smem[off];
      b.h2[1] = *(const v8h*)&smem[off + 8];
    } else {                                     // IC == 4 (input layer)
#pragma unroll
      for (int q = 0; q < 4; ++q) {
        const int c0 = boff4<IC, KH>(chunk, 0, q);
        const int c1 = boff4<IC, KH>(chunk, 1, q);
        const int o0 = (c0 < 0) ? SW_Z : (pxIC + c0);
        const int o1 = (c1 < 0) ? SW_Z : (pxIC + c1);
        const int off = grp ? o1 : o0;
        b.q4[q] = *(const v4h*)&smem[off];
      }
    }
    // ---- A fragments (block-uniform) + WMMA per M-tile ----
#pragma unroll
    for (int mt = 0; mt < MT; ++mt) {
      union { v16h v; v8h h2[2]; } a;
      if constexpr (STW) {
        const int ao = (mt * KC + chunk) * 512 + lane16;
        a.h2[0] = *(const v8h*)&smem[ao];
        a.h2[1] = *(const v8h*)&smem[ao + 8];
      } else {
        const _Float16* ap = &wl[(size_t)((mt * KC + chunk) * 32) * 16 + lane16];
        a.h2[0] = *(const v8h*)ap;
        a.h2[1] = *(const v8h*)(ap + 8);
      }
      acc[mt] = __builtin_amdgcn_wmma_f32_16x16x32_f16(
          false, a.v, false, b.v, (short)0, acc[mt], false, false);
    }
  }

  // ---- epilogue: bias + tanh, 16B NHWC store per acc ----
#pragma unroll
  for (int mt = 0; mt < MT; ++mt) {
    const int oc0 = mt * 16 + grp * 8;
    const float* bp = bias + (size_t)sub * OC + oc0;
    v8h o;
#pragma unroll
    for (int r = 0; r < 8; ++r)
      o[r] = (_Float16)tanh_fast(acc[mt][r] + bp[r]);
    *(v8h*)&out[(((size_t)img * HH + y) * WW + px) * OC + oc0] = o;
  }
}

// ---------------------------------------------------------------------------
// Layer 5 (1x1, 16->3) fused with masked Bernoulli log-prob + reduction.
//   lp = -softplus(-z) - (1-s)*z
// ---------------------------------------------------------------------------
__global__ __launch_bounds__(256)
void l5_logprob(const _Float16* __restrict__ h4,
                const float* __restrict__ w5, const float* __restrict__ b5,
                const float* __restrict__ samples, float* __restrict__ out) {
  __shared__ float red[256];
  const int img = blockIdx.x;               // 0..191
  const int n = img >> 4, b = img & 15;
  const int p = n / 3, c = n % 3;
  const int yp = (p == 1 || p == 3) ? 1 : 0;
  const int xp = (p == 1 || p == 2) ? 1 : 0;
  float wl[16];
#pragma unroll
  for (int ic = 0; ic < 16; ++ic) wl[ic] = w5[(n * 3 + c) * 16 + ic];
  const float bz = b5[n * 3 + c];

  float acc = 0.0f;
  for (int i = threadIdx.x; i < 4096; i += 256) {
    const int y = ((i >> 6) << 1) + yp;
    const int x = ((i & 63) << 1) + xp;
    const _Float16* hp = h4 + (((size_t)img * HH + y) * WW + x) * 16;
    float z = bz;
#pragma unroll
    for (int ic = 0; ic < 16; ++ic) z += wl[ic] * (float)hp[ic];
    const float s  = samples[(((b * 3 + c) * HH) + y) * WW + x];
    const float t  = __builtin_amdgcn_exp2f(-z * 1.4426950408889634f);
    const float sp = 0.6931471805599453f * __builtin_amdgcn_logf(1.0f + t);
    acc += -sp - (1.0f - s) * z;
  }
  red[threadIdx.x] = acc;
  __syncthreads();
  for (int s2 = 128; s2 > 0; s2 >>= 1) {
    if (threadIdx.x < s2) red[threadIdx.x] += red[threadIdx.x + s2];
    __syncthreads();
  }
  if (threadIdx.x == 0) atomicAdd(&out[b], red[0]);
}

__global__ void zero_out(float* __restrict__ o) {
  if (threadIdx.x < BATCH) o[threadIdx.x] = 0.0f;
}

// ---------------------------------------------------------------------------
extern "C" void kernel_launch(void* const* d_in, const int* in_sizes, int n_in,
                              void* d_out, int out_size, void* d_ws, size_t ws_size,
                              hipStream_t stream) {
  (void)in_sizes; (void)n_in; (void)out_size; (void)ws_size;
  const float* samples = (const float*)d_in[0];
  const float* cond    = (const float*)d_in[1];
  const float* w0 = (const float*)d_in[2];  const float* b0 = (const float*)d_in[3];
  const float* w1 = (const float*)d_in[4];  const float* b1 = (const float*)d_in[5];
  const float* w2 = (const float*)d_in[6];  const float* b2 = (const float*)d_in[7];
  const float* w3 = (const float*)d_in[8];  const float* b3 = (const float*)d_in[9];
  const float* w4 = (const float*)d_in[10]; const float* b4 = (const float*)d_in[11];
  const float* w5 = (const float*)d_in[12]; const float* b5 = (const float*)d_in[13];

  // Workspace layout (f16 halves). Packed weights ~1MB, activations ~0.93GB.
  _Float16* wp0 = (_Float16*)d_ws;
  _Float16* wp1 = wp0 + (size_t)NSUB * 1 * 2  * 512;   // L0: MT=1,KC=2
  _Float16* wp2 = wp1 + (size_t)NSUB * 2 * 5  * 512;   // L1: MT=2,KC=5
  _Float16* wp3 = wp2 + (size_t)NSUB * 4 * 9  * 512;   // L2: MT=4,KC=9
  _Float16* wp4 = wp3 + (size_t)NSUB * 2 * 18 * 512;   // L3: MT=2,KC=18
  _Float16* xin = wp4 + (size_t)NSUB * 1 * 1  * 512;   // L4: MT=1,KC=1
  const size_t PIX = (size_t)NB * HH * WW;             // 3,145,728 pixels
  _Float16* a0 = xin + PIX * 4;    // 16-ch
  _Float16* a1 = a0  + PIX * 16;   // 32-ch
  _Float16* a2 = a1  + PIX * 32;   // 64-ch
  _Float16* a3 = a2  + PIX * 64;   // 32-ch
  _Float16* a4 = a0;               // 16-ch, reuses dead act0 region

  zero_out<<<1, 32, 0, stream>>>((float*)d_out);

  pack_w<4, 16, 3, 2 ><<<(NSUB*1*2 *512 + 255)/256, 256, 0, stream>>>(w0, wp0);
  pack_w<16,32, 3, 5 ><<<(NSUB*2*5 *512 + 255)/256, 256, 0, stream>>>(w1, wp1);
  pack_w<32,64, 3, 9 ><<<(NSUB*4*9 *512 + 255)/256, 256, 0, stream>>>(w2, wp2);
  pack_w<64,32, 3, 18><<<(NSUB*2*18*512 + 255)/256, 256, 0, stream>>>(w3, wp3);
  pack_w<32,16, 1, 1 ><<<(NSUB*1*1 *512 + 255)/256, 256, 0, stream>>>(w4, wp4);

  prep_input<<<(int)((PIX + 255) / 256), 256, 0, stream>>>(samples, cond, xin);

  const dim3 cg(NB * HH);   // one block per (image, row)
  conv_wmma<4, 16, 3, 2 ><<<cg, 256, 0, stream>>>(xin, wp0, b0, a0);
  conv_wmma<16,32, 3, 5 ><<<cg, 256, 0, stream>>>(a0,  wp1, b1, a1);
  conv_wmma<32,64, 3, 9 ><<<cg, 256, 0, stream>>>(a1,  wp2, b2, a2);
  conv_wmma<64,32, 3, 18><<<cg, 256, 0, stream>>>(a2,  wp3, b3, a3);
  conv_wmma<32,16, 1, 1 ><<<cg, 256, 0, stream>>>(a3,  wp4, b4, a4);

  l5_logprob<<<NB, 256, 0, stream>>>(a4, w5, b5, samples, (float*)d_out);
}
